// CrystalMancerGNN_65146063946419
// MI455X (gfx1250) — matrix operand order repaired and checked by reading, not compiled, not run on hardware
//
#include <hip/hip_runtime.h>
#include <hip/hip_bf16.h>

// ---------------- CDNA5 types ----------------
typedef __attribute__((ext_vector_type(16))) __bf16 v16bf;
typedef __attribute__((ext_vector_type(8)))  __bf16 v8bf;
typedef __attribute__((ext_vector_type(8)))  float  v8f;

#define NN  50000
#define EE  800000
#define BB  256
#define HH  128
#define LL  4
#define AF  108
#define EF  41
#define GF  239
#define TT  5

__device__ __forceinline__ __bf16 f2bf(float f) {
    union { float f; unsigned u; } in; in.f = f;
    unsigned u = in.u;
    unsigned r = (u + 0x7FFFu + ((u >> 16) & 1u)) >> 16;   // RNE
    union { unsigned short s; __bf16 b; } out; out.s = (unsigned short)r;
    return out.b;
}

// silu via hardware v_exp_f32 + v_rcp_f32 (avoids the IEEE div ladder)
__device__ __forceinline__ float siluf(float x) {
    return x * __builtin_amdgcn_rcpf(1.f + __expf(-x));
}

// One 16x16 output tile: D = A(16xK bf16, LDS row-major) @ Wp + C
// Wp is PRE-PACKED in B-fragment layout: for this column block,
//   Wp[kb*512 + lane*16 + j] == W[kb*32 + 16*(lane>>4) + j][n0 + (lane&15)]
// so each lane's fragment is one contiguous 32-byte load.
template <int KB>   // number of 32-wide k blocks
__device__ __forceinline__ v8f wmma_tile(const __bf16* __restrict__ ldsA, int lda,
                                         const __bf16* __restrict__ Wp,
                                         int lane, v8f c) {
    const int lo = lane & 15;
    const int hi = lane >> 4;
    const __bf16* bp = Wp + lane * 16;
#pragma unroll
    for (int kb = 0; kb < KB; ++kb) {
        const __bf16* ap = ldsA + lo * lda + kb * 32 + 16 * hi;
        v8bf alo = *(const v8bf*)ap;          // K = kb*32 + 16*hi + [0..7]
        v8bf ahi = *(const v8bf*)(ap + 8);    // K = kb*32 + 16*hi + [8..15]
        v16bf a = __builtin_shufflevector(alo, ahi, 0, 1, 2, 3, 4, 5, 6, 7,
                                          8, 9, 10, 11, 12, 13, 14, 15);
        v16bf b = *(const v16bf*)(bp + kb * 512);   // 32B contiguous per lane
        c = __builtin_amdgcn_wmma_f32_16x16x32_bf16(false, a, false, b, (short)0, c,
                                                    false, false);
    }
    return c;
}

// ---------------- weight fp32 -> packed bf16 fragments (zero K-padded) ----------------
// dst layout: idx = ((m*NB + cb)*KB + kb)*512 + lane*16 + j
__global__ void cvt_pack_kernel(const float* __restrict__ src, __bf16* __restrict__ dst,
                                int K, int KB, int Ncols, int total) {
    int idx = blockIdx.x * blockDim.x + threadIdx.x;
    if (idx >= total) return;
    int j    = idx & 15;
    int lane = (idx >> 4) & 31;
    int blk  = idx >> 9;
    int kb   = blk % KB;
    int rest = blk / KB;
    int NB   = Ncols >> 4;
    int cb   = rest % NB;
    int m    = rest / NB;
    int k = kb * 32 + ((lane >> 4) << 4) + j;
    int n = cb * 16 + (lane & 15);
    float v = (k < K) ? src[((size_t)m * K + k) * Ncols + n] : 0.f;
    dst[idx] = f2bf(v);
}

// ---------------- atom embedding: h = silu(x@W1+b1)@W2+b2 ----------------
__global__ void embed_kernel(const float* __restrict__ x,
                             const __bf16* __restrict__ w1, const float* __restrict__ b1,
                             const __bf16* __restrict__ w2, const float* __restrict__ b2,
                             float* __restrict__ h) {
    __shared__ __align__(16) __bf16 lds_x[16 * 128];
    __shared__ __align__(16) __bf16 lds_t[16 * 128];
    const int tid = threadIdx.x;
    const int n0b = blockIdx.x * 16;
    for (int i = tid; i < 16 * 128; i += 256) {
        int r = i >> 7, c = i & 127;
        float v = (c < AF) ? x[(size_t)(n0b + r) * AF + c] : 0.f;
        lds_x[i] = f2bf(v);
    }
    __syncthreads();
    const int w = tid >> 5, lane = tid & 31, lo = lane & 15, hi = lane >> 4, n0 = w * 16;
    v8f c1 = {};
    c1 = wmma_tile<4>(lds_x, 128, w1 + w * 4 * 512, lane, c1);
    float bias1 = b1[n0 + lo];
#pragma unroll
    for (int r = 0; r < 8; ++r) {
        int row = r + 8 * hi;
        lds_t[row * 128 + n0 + lo] = f2bf(siluf(c1[r] + bias1));
    }
    __syncthreads();
    v8f c2 = {};
    c2 = wmma_tile<4>(lds_t, 128, w2 + w * 4 * 512, lane, c2);
    float bias2 = b2[n0 + lo];
#pragma unroll
    for (int r = 0; r < 8; ++r) {
        int row = r + 8 * hi;
        h[(size_t)(n0b + row) * HH + n0 + lo] = c2[r] + bias2;
    }
}

// ---------------- fused edge layer ----------------
// Wf = silu(ea@W1+b1)@W2+b2 ; msg = h[src]*Wf ; atomic scatter into agg[dst]
__global__ void edge_kernel(const float* __restrict__ ea, const int* __restrict__ eidx,
                            const float* __restrict__ h,
                            const __bf16* __restrict__ w1, const float* __restrict__ b1,
                            const __bf16* __restrict__ w2, const float* __restrict__ b2,
                            float* __restrict__ agg) {
    __shared__ __align__(16) __bf16 lds_ea[16 * 64];
    __shared__ __align__(16) __bf16 lds_t[16 * 128];
    __shared__ __align__(16) float  lds_hs[16 * 128];
    const int tid = threadIdx.x;
    const int e0 = blockIdx.x * 16;
    for (int i = tid; i < 16 * 64; i += 256) {
        int r = i >> 6, c = i & 63;
        float v = (c < EF) ? ea[(size_t)(e0 + r) * EF + c] : 0.f;
        lds_ea[i] = f2bf(v);
    }
    // gather h[src] rows, 16B vectorized
    for (int i = tid; i < 16 * 32; i += 256) {
        int r = i >> 5, c4 = i & 31;
        int s = eidx[e0 + r];
        ((float4*)lds_hs)[i] = ((const float4*)(h + (size_t)s * HH))[c4];
    }
    __syncthreads();
    const int w = tid >> 5, lane = tid & 31, lo = lane & 15, hi = lane >> 4, n0 = w * 16;
    v8f c1 = {};
    c1 = wmma_tile<2>(lds_ea, 64, w1 + w * 2 * 512, lane, c1);
    float bias1 = b1[n0 + lo];
#pragma unroll
    for (int r = 0; r < 8; ++r) {
        int row = r + 8 * hi;
        lds_t[row * 128 + n0 + lo] = f2bf(siluf(c1[r] + bias1));
    }
    __syncthreads();
    v8f c2 = {};
    c2 = wmma_tile<4>(lds_t, 128, w2 + w * 4 * 512, lane, c2);
    float bias2 = b2[n0 + lo];
#pragma unroll
    for (int r = 0; r < 8; ++r) {
        int row = r + 8 * hi;
        int col = n0 + lo;
        float wf = c2[r] + bias2;
        float msg = lds_hs[row * 128 + col] * wf;
        int d = eidx[EE + e0 + row];                // dst node
        atomicAdd(&agg[(size_t)d * HH + col], msg); // global_atomic_add_f32
    }
}

// ---------------- fused node layer: out = silu(agg@W1+b1)@W2+b2 ; h = LN(h+out) ----------------
__global__ void node_kernel(const float* __restrict__ agg, float* __restrict__ h,
                            const __bf16* __restrict__ w1, const float* __restrict__ b1,
                            const __bf16* __restrict__ w2, const float* __restrict__ b2,
                            const float* __restrict__ lng, const float* __restrict__ lnb) {
    __shared__ __align__(16) __bf16 lds_a[16 * 128];
    __shared__ __align__(16) __bf16 lds_t[16 * 128];
    __shared__ __align__(16) float  lds_o[16 * 128];
    const int tid = threadIdx.x;
    const int n0b = blockIdx.x * 16;
    for (int i = tid; i < 16 * 32; i += 256) {
        float4 v = ((const float4*)(agg + (size_t)n0b * HH))[i];
        __bf16* d = lds_a + i * 4;
        d[0] = f2bf(v.x); d[1] = f2bf(v.y); d[2] = f2bf(v.z); d[3] = f2bf(v.w);
    }
    __syncthreads();
    const int w = tid >> 5, lane = tid & 31, lo = lane & 15, hi = lane >> 4, n0 = w * 16;
    v8f c1 = {};
    c1 = wmma_tile<4>(lds_a, 128, w1 + w * 4 * 512, lane, c1);
    float bias1 = b1[n0 + lo];
#pragma unroll
    for (int r = 0; r < 8; ++r) {
        int row = r + 8 * hi;
        lds_t[row * 128 + n0 + lo] = f2bf(siluf(c1[r] + bias1));
    }
    __syncthreads();
    v8f c2 = {};
    c2 = wmma_tile<4>(lds_t, 128, w2 + w * 4 * 512, lane, c2);
    float bias2 = b2[n0 + lo];
#pragma unroll
    for (int r = 0; r < 8; ++r) {
        int row = r + 8 * hi;
        lds_o[row * 128 + n0 + lo] = c2[r] + bias2;
    }
    __syncthreads();
    // residual + LayerNorm: wave w handles rows 2w, 2w+1; 4 elems/lane, wave32 shuffle reduce
    for (int rr = 0; rr < 2; ++rr) {
        int row = w * 2 + rr;
        size_t node = (size_t)(n0b + row);
        float v[4];
        float s = 0.f;
#pragma unroll
        for (int j = 0; j < 4; ++j) {
            int col = lane * 4 + j;
            v[j] = h[node * HH + col] + lds_o[row * 128 + col];
            s += v[j];
        }
        for (int m = 16; m >= 1; m >>= 1) s += __shfl_xor(s, m, 32);
        float mean = s * (1.f / 128.f);
        float q = 0.f;
#pragma unroll
        for (int j = 0; j < 4; ++j) { float d = v[j] - mean; q += d * d; }
        for (int m = 16; m >= 1; m >>= 1) q += __shfl_xor(q, m, 32);
        float rstd = rsqrtf(q * (1.f / 128.f) + 1e-5f);
#pragma unroll
        for (int j = 0; j < 4; ++j) {
            int col = lane * 4 + j;
            h[node * HH + col] = (v[j] - mean) * rstd * lng[col] + lnb[col];
        }
    }
}

// ---------------- head ----------------
__global__ void count_kernel(const int* __restrict__ batch, float* __restrict__ cnt) {
    int i = blockIdx.x * blockDim.x + threadIdx.x;
    if (i < NN) atomicAdd(&cnt[batch[i]], 1.f);
}

__global__ void pool_sum_kernel(const float* __restrict__ h, const int* __restrict__ batch,
                                float* __restrict__ wsum) {
    int idx = blockIdx.x * blockDim.x + threadIdx.x;
    if (idx >= NN * HH) return;
    int i = idx >> 7, c = idx & 127;
    atomicAdd(&wsum[batch[i] * HH + c], h[idx]);
}

__global__ void comb_kernel(const float* __restrict__ wsum, const float* __restrict__ cnt,
                            const float* __restrict__ gfeat, const float* __restrict__ gw,
                            const float* __restrict__ gb, float* __restrict__ comb) {
    int idx = blockIdx.x * blockDim.x + threadIdx.x;   // BB * 2H
    if (idx >= BB * 2 * HH) return;
    int b = idx >> 8, c = idx & 255;
    float v;
    if (c < HH) {
        v = wsum[b * HH + c] * __builtin_amdgcn_rcpf(fmaxf(cnt[b], 1.f));
    } else {
        int n = c - HH;
        float acc = gb[n];
        for (int g = 0; g < GF; ++g) acc += gfeat[b * GF + g] * gw[g * HH + n];
        v = siluf(acc);
    }
    comb[idx] = v;
}

__global__ void dense_kernel(const float* __restrict__ X, const float* __restrict__ W,
                             const float* __restrict__ bias, float* __restrict__ Y,
                             int M, int K, int Nout, int act) {
    int idx = blockIdx.x * blockDim.x + threadIdx.x;
    if (idx >= M * Nout) return;
    int m = idx / Nout, n = idx % Nout;
    float acc = bias[n];
    for (int k = 0; k < K; ++k) acc += X[m * K + k] * W[k * Nout + n];
    Y[idx] = act ? siluf(acc) : acc;
}

// ---------------- launch ----------------
extern "C" void kernel_launch(void* const* d_in, const int* in_sizes, int n_in,
                              void* d_out, int out_size, void* d_ws, size_t ws_size,
                              hipStream_t stream) {
    const float* x      = (const float*)d_in[0];
    const float* ea     = (const float*)d_in[1];
    const float* gfeat  = (const float*)d_in[2];
    const float* ae_w1  = (const float*)d_in[3];
    const float* ae_b1  = (const float*)d_in[4];
    const float* ae_w2  = (const float*)d_in[5];
    const float* ae_b2  = (const float*)d_in[6];
    const float* ew1    = (const float*)d_in[7];
    const float* eb1    = (const float*)d_in[8];
    const float* ew2    = (const float*)d_in[9];
    const float* eb2    = (const float*)d_in[10];
    const float* nw1    = (const float*)d_in[11];
    const float* nb1    = (const float*)d_in[12];
    const float* nw2    = (const float*)d_in[13];
    const float* nb2    = (const float*)d_in[14];
    const float* ln_g   = (const float*)d_in[15];
    const float* ln_b   = (const float*)d_in[16];
    const float* gw     = (const float*)d_in[17];
    const float* gb     = (const float*)d_in[18];
    const float* ow1    = (const float*)d_in[19];
    const float* ob1    = (const float*)d_in[20];
    const float* ow2    = (const float*)d_in[21];
    const float* ob2    = (const float*)d_in[22];
    const float* ow3    = (const float*)d_in[23];
    const float* ob3    = (const float*)d_in[24];
    const int*   eidx   = (const int*)d_in[25];
    const int*   batch  = (const int*)d_in[26];
    float* out = (float*)d_out;

    // workspace layout (floats)
    float* ws = (float*)d_ws;
    size_t o = 0;
    float* agg  = ws + o; o += (size_t)NN * HH;
    float* hbuf = ws + o; o += (size_t)NN * HH;
    float* wsum = ws + o; o += BB * HH;
    float* cnt  = ws + o; o += BB;
    float* comb = ws + o; o += BB * 2 * HH;
    float* hid1 = ws + o; o += BB * HH;
    float* hid2 = ws + o; o += BB * 64;
    __bf16* bf = (__bf16*)(ws + o);
    size_t bo = 0;
    // packed sizes: NB * KB * 512 bf16 per matrix (NB = Ncols/16, KB = Kpad/32)
    __bf16* ae1b = bf + bo; bo += 8 * 4 * 512;               // 108->128
    __bf16* ae2b = bf + bo; bo += 8 * 4 * 512;               // 128
    __bf16* ew1b = bf + bo; bo += (size_t)LL * 8 * 2 * 512;  // 41->64
    __bf16* ew2b = bf + bo; bo += (size_t)LL * 8 * 4 * 512;  // 128
    __bf16* nw1b = bf + bo; bo += (size_t)LL * 8 * 4 * 512;  // 128
    __bf16* nw2b = bf + bo; bo += (size_t)LL * 8 * 4 * 512;  // 128

    // 1) convert weights to packed bf16 fragments (zero K-padded)
    cvt_pack_kernel<<<(8 * 4 * 512 + 255) / 256, 256, 0, stream>>>(ae_w1, ae1b, AF, 4, 128, 8 * 4 * 512);
    cvt_pack_kernel<<<(8 * 4 * 512 + 255) / 256, 256, 0, stream>>>(ae_w2, ae2b, 128, 4, 128, 8 * 4 * 512);
    cvt_pack_kernel<<<(LL * 8 * 2 * 512 + 255) / 256, 256, 0, stream>>>(ew1, ew1b, EF, 2, 128, LL * 8 * 2 * 512);
    cvt_pack_kernel<<<(LL * 8 * 4 * 512 + 255) / 256, 256, 0, stream>>>(ew2, ew2b, 128, 4, 128, LL * 8 * 4 * 512);
    cvt_pack_kernel<<<(LL * 8 * 4 * 512 + 255) / 256, 256, 0, stream>>>(nw1, nw1b, 128, 4, 128, LL * 8 * 4 * 512);
    cvt_pack_kernel<<<(LL * 8 * 4 * 512 + 255) / 256, 256, 0, stream>>>(nw2, nw2b, 128, 4, 128, LL * 8 * 4 * 512);

    // 2) atom embedding
    embed_kernel<<<NN / 16, 256, 0, stream>>>(x, ae1b, ae_b1, ae2b, ae_b2, hbuf);

    // 3) message-passing layers (fused edge MLP + gather/scatter, fused node MLP + LN)
    for (int l = 0; l < LL; ++l) {
        hipMemsetAsync(agg, 0, (size_t)NN * HH * sizeof(float), stream);
        edge_kernel<<<EE / 16, 256, 0, stream>>>(ea, eidx, hbuf,
                                                 ew1b + (size_t)l * 8 * 2 * 512, eb1 + l * HH,
                                                 ew2b + (size_t)l * 8 * 4 * 512, eb2 + l * HH,
                                                 agg);
        node_kernel<<<NN / 16, 256, 0, stream>>>(agg, hbuf,
                                                 nw1b + (size_t)l * 8 * 4 * 512, nb1 + l * HH,
                                                 nw2b + (size_t)l * 8 * 4 * 512, nb2 + l * HH,
                                                 ln_g + l * HH, ln_b + l * HH);
    }

    // 4) pooling + head
    hipMemsetAsync(wsum, 0, (BB * HH + BB) * sizeof(float), stream);  // wsum + cnt contiguous
    count_kernel<<<(NN + 255) / 256, 256, 0, stream>>>(batch, cnt);
    pool_sum_kernel<<<(NN * HH + 255) / 256, 256, 0, stream>>>(hbuf, batch, wsum);
    comb_kernel<<<(BB * 2 * HH + 255) / 256, 256, 0, stream>>>(wsum, cnt, gfeat, gw, gb, comb);
    dense_kernel<<<(BB * HH + 255) / 256, 256, 0, stream>>>(comb, ow1, ob1, hid1, BB, 2 * HH, HH, 1);
    dense_kernel<<<(BB * 64 + 255) / 256, 256, 0, stream>>>(hid1, ow2, ob2, hid2, BB, HH, 64, 1);
    dense_kernel<<<(BB * TT + 255) / 256, 256, 0, stream>>>(hid2, ow3, ob3, out, BB, 64, TT, 0);

    (void)in_sizes; (void)n_in; (void)out_size; (void)ws_size;
}
